// IGMC_26637387170068
// MI455X (gfx1250) — matrix operand order repaired and checked by reading, not compiled, not run on hardware
//
#include <hip/hip_runtime.h>

// ---------------- problem constants (match reference) ----------------
#define N_NODES 131072
#define N_EDGES 2097152
#define BATCH   4096
#define H_DIM   32
#define N_LAYERS 4
#define N_REL   5
#define N_BASES 4
#define D_LSTM  256
#define G_LSTM  1024   // 4*D

// ---------------- WMMA plumbing (gfx1250, wave32) ----------------
typedef __attribute__((ext_vector_type(16))) __bf16 v16bf;
typedef __attribute__((ext_vector_type(8)))  __bf16 v8bf;
typedef __attribute__((ext_vector_type(8)))  float  v8f;

__device__ __forceinline__ v8f wmma_bf16(v16bf a, v16bf b, v8f c) {
  // v_wmma_f32_16x16x32_bf16  (D = A(16x32) * B(32x16) + C)
  return __builtin_amdgcn_wmma_f32_16x16x32_bf16(
      /*neg_a=*/false, a, /*neg_b=*/false, b,
      /*c_mod=*/(short)0, c, /*reuse_a=*/false, /*reuse_b=*/false);
}

__device__ __forceinline__ v16bf combine16(v8bf lo, v8bf hi) {
  return __builtin_shufflevector(lo, hi, 0, 1, 2, 3, 4, 5, 6, 7,
                                          8, 9, 10, 11, 12, 13, 14, 15);
}

// A 16x32 tile, row-major source with leading dim lda (elements).
// Per-lane data = one contiguous 16-element run: k in [16*half, +16).
// -> two global_load_b128 per lane.
__device__ __forceinline__ v16bf load_A_16x32(const __bf16* A, int lda) {
  int lane = threadIdx.x & 31;
  int row  = lane & 15;
  int half = lane >> 4;
  const __bf16* p = A + (size_t)row * lda + 16 * half;
  v8bf lo = *(const v8bf*)(p);        // k = 16*half .. +7
  v8bf hi = *(const v8bf*)(p + 8);    // k = 16*half+8 .. +7
  return combine16(lo, hi);
}

// B 32x16 tile where the source is N x K row-major (weight rows = output cols):
// B[k][n] = W[(n0+n)*ldw + k0+k].  Per-lane data = one contiguous 16-element run.
__device__ __forceinline__ v16bf load_B_trans(const __bf16* W, int ldw, int n0, int k0) {
  int lane = threadIdx.x & 31;
  int n    = lane & 15;
  int half = lane >> 4;
  const __bf16* p = W + (size_t)(n0 + n) * ldw + k0 + 16 * half;
  v8bf lo = *(const v8bf*)(p);
  v8bf hi = *(const v8bf*)(p + 8);
  return combine16(lo, hi);
}

// B fragment pre-swizzled in memory: F[lane][0..15] contiguous (32 B per lane).
__device__ __forceinline__ v16bf load_B_frag(const __bf16* F) {
  int lane = threadIdx.x & 31;
  const __bf16* p = F + lane * 16;
  v8bf lo = *(const v8bf*)(p);
  v8bf hi = *(const v8bf*)(p + 8);
  return combine16(lo, hi);
}

__device__ __forceinline__ float sigmoidf(float x) { return 1.0f / (1.0f + __expf(-x)); }

// ---------------- kernels ----------------

__global__ void cast_f32_bf16(const float* __restrict__ src, __bf16* __restrict__ dstp, int n) {
  int i = blockIdx.x * blockDim.x + threadIdx.x;
  if (i < n) dstp[i] = (__bf16)src[i];
}

// per-(dst,relation) edge counts for mean normalization
__global__ void edge_count(const int* __restrict__ dst, const int* __restrict__ etype,
                           unsigned* __restrict__ counts) {
  int e = blockIdx.x * blockDim.x + threadIdx.x;
  if (e < N_EDGES) atomicAdd(&counts[dst[e] * N_REL + etype[e]], 1u);
}

__global__ void edge_norm_k(const int* __restrict__ dst, const int* __restrict__ etype,
                            const unsigned* __restrict__ counts, float* __restrict__ enorm) {
  int e = blockIdx.x * blockDim.x + threadIdx.x;
  if (e < N_EDGES) {
    unsigned c = counts[dst[e] * N_REL + etype[e]];
    enorm[e] = 1.0f / (float)(c > 1u ? c : 1u);
  }
}

// Build per-layer relation weights W_r = sum_b comp[l,r,b]*basis[l,b]  (+ root as slot R),
// emitted directly in WMMA B-fragment-swizzled bf16 layout:
//   Wb[l][m][half][lane][j]  with  k = j + 16*(lane>>4),  n = half*16 + (lane&15)
// so each lane's 16 values are contiguous (two b128 loads at use site).
__global__ void build_relw(const float* __restrict__ basis, const float* __restrict__ comp,
                           const float* __restrict__ root, __bf16* __restrict__ Wb) {
  int t = blockIdx.x * blockDim.x + threadIdx.x;       // L*6*2*32*16 = 24576
  if (t >= N_LAYERS * (N_REL + 1) * 2 * 32 * 16) return;
  int j    = t & 15;
  int lane = (t >> 4) & 31;
  int half = (t >> 9) & 1;
  int m    = (t >> 10) % (N_REL + 1);
  int l    = t / ((N_REL + 1) * 2 * 32 * 16);
  int k = j + 16 * (lane >> 4);        // input dim d
  int n = half * 16 + (lane & 15);     // output dim e
  float w;
  if (m < N_REL) {
    w = 0.0f;
#pragma unroll
    for (int b = 0; b < N_BASES; ++b)
      w += comp[(l * N_REL + m) * N_BASES + b] *
           basis[(((l * N_BASES + b) * H_DIM) + k) * H_DIM + n];
  } else {
    w = root[(l * H_DIM + k) * H_DIM + n];
  }
  Wb[t] = (__bf16)w;
}

// WMMA transform: for a 16-node tile, y[n,r,:] = h[n] @ W_r  and  agg[n,:] = h[n] @ root + bias.
// One wave per tile; 6 matrices x 2 column halves = 12 WMMAs, K=32 single step.
__global__ void rgcn_transform(const __bf16* __restrict__ h_bf, const __bf16* __restrict__ Wb_l,
                               const float* __restrict__ bias_l,
                               float* __restrict__ y, float* __restrict__ agg) {
  int wave = (blockIdx.x * blockDim.x + threadIdx.x) >> 5;   // tile index (wave-uniform)
  if (wave * 16 >= N_NODES) return;
  int lane  = threadIdx.x & 31;
  int ncolL = lane & 15;
  int mbase = (lane >> 4) * 8;
  int row0  = wave * 16;

  v16bf a = load_A_16x32(h_bf + (size_t)row0 * H_DIM, H_DIM);

#pragma unroll
  for (int m = 0; m <= N_REL; ++m) {
#pragma unroll
    for (int half = 0; half < 2; ++half) {
      v16bf b = load_B_frag(Wb_l + ((m * 2 + half) * 32) * 16);
      v8f c = {};
      c = wmma_bf16(a, b, c);
      int ncol = ncolL + half * 16;
      if (m < N_REL) {
#pragma unroll
        for (int i = 0; i < 8; ++i)
          y[((size_t)(row0 + mbase + i) * N_REL + m) * H_DIM + ncol] = c[i];
      } else {
        float bv = bias_l[ncol];
#pragma unroll
        for (int i = 0; i < 8; ++i)
          agg[(size_t)(row0 + mbase + i) * H_DIM + ncol] = c[i] + bv;
      }
    }
  }
}

// Per-edge scatter: agg[dst] += y[src, rel] * norm.  8 threads/edge, float4 gather + 4 f32 atomics.
__global__ void rgcn_scatter(const int* __restrict__ src, const int* __restrict__ dst,
                             const int* __restrict__ etype, const float* __restrict__ enorm,
                             const float* __restrict__ y, float* __restrict__ agg) {
  int t = blockIdx.x * blockDim.x + threadIdx.x;     // E*8 = 16.7M threads
  int e = t >> 3;
  int q = t & 7;
  if (e >= N_EDGES) return;
  float nv = enorm[e];
  const float4 v = *(const float4*)(y + ((size_t)src[e] * N_REL + etype[e]) * H_DIM + q * 4);
  float* ap = agg + (size_t)dst[e] * H_DIM + q * 4;
  atomicAdd(ap + 0, v.x * nv);
  atomicAdd(ap + 1, v.y * nv);
  atomicAdd(ap + 2, v.z * nv);
  atomicAdd(ap + 3, v.w * nv);
}

// h = tanh(agg); write into states[:, l*32 .. l*32+31] and refresh bf16 h for next layer.
__global__ void rgcn_update(const float* __restrict__ agg, float* __restrict__ states,
                            __bf16* __restrict__ h_bf, int l) {
  int i = blockIdx.x * blockDim.x + threadIdx.x;
  if (i >= N_NODES * H_DIM) return;
  int node = i >> 5, c = i & 31;
  float v = tanhf(agg[i]);
  states[(size_t)node * 128 + l * H_DIM + c] = v;
  h_bf[i] = (__bf16)v;
}

// z[i] = concat(states[user[i]], states[item[i]]) as bf16 for the x-gates GEMM.
__global__ void gather_z(const float* __restrict__ states, const int* __restrict__ uidx,
                         const int* __restrict__ vidx, __bf16* __restrict__ zb) {
  int t = blockIdx.x * blockDim.x + threadIdx.x;     // B*256
  if (t >= BATCH * D_LSTM) return;
  int i = t >> 8, c = t & 255;
  float v = (c < 128) ? states[(size_t)uidx[i] * 128 + c]
                      : states[(size_t)vidx[i] * 128 + (c - 128)];
  zb[t] = (__bf16)v;
}

// x_gates = z @ w_ih^T + b_ih + b_hh : WMMA GEMM M=4096, N=1024, K=256 (8 k-steps).
__global__ void lstm_xgates(const __bf16* __restrict__ zin, const __bf16* __restrict__ wih_bf,
                            const float* __restrict__ b_ih_l, const float* __restrict__ b_hh_l,
                            float* __restrict__ xg) {
  int wave = (blockIdx.x * blockDim.x + threadIdx.x) >> 5;   // 16384 tiles (wave-uniform)
  int mt = wave >> 6;         // 256 row tiles
  int nt = wave & 63;         // 64 gate tiles
  if (mt >= BATCH / 16) return;
  v8f c = {};
#pragma unroll
  for (int ks = 0; ks < D_LSTM / 32; ++ks) {
    v16bf a = load_A_16x32(zin + (size_t)mt * 16 * D_LSTM + ks * 32, D_LSTM);
    v16bf b = load_B_trans(wih_bf, D_LSTM, nt * 16, ks * 32);
    c = wmma_bf16(a, b, c);
  }
  int lane  = threadIdx.x & 31;
  int g     = nt * 16 + (lane & 15);
  int mbase = (lane >> 4) * 8;
  float bias = b_ih_l[g] + b_hh_l[g];
#pragma unroll
  for (int i = 0; i < 8; ++i)
    xg[(size_t)(mt * 16 + mbase + i) * G_LSTM + g] = c[i] + bias;
}

// Sequential LSTM scan: one 1024-thread workgroup (32 wave32s on one WGP).
// h/gates in LDS, per-thread c-state in a register, w_hh streams from L2 (1 MB/step).
__global__ void lstm_scan(const float* __restrict__ xg, const float* __restrict__ w_hh_l,
                          float* __restrict__ outf, __bf16* __restrict__ outb) {
  __shared__ float hs[D_LSTM];
  __shared__ float gates[G_LSTM];
  int j = threadIdx.x;                       // 0..1023, one gate per thread
  float cst = 0.0f;
  if (j < D_LSTM) hs[j] = 0.0f;
  __syncthreads();
  const float4* wr = (const float4*)(w_hh_l + (size_t)j * D_LSTM);
  for (int t = 0; t < BATCH; ++t) {
    const float4* hv = (const float4*)hs;
    float acc = xg[(size_t)t * G_LSTM + j];
#pragma unroll 8
    for (int k = 0; k < D_LSTM / 4; ++k) {
      float4 w = wr[k];
      float4 h4 = hv[k];
      acc += w.x * h4.x + w.y * h4.y + w.z * h4.z + w.w * h4.w;
    }
    gates[j] = acc;
    __syncthreads();
    if (j < D_LSTM) {
      float ig = sigmoidf(gates[j]);
      float fg = sigmoidf(gates[D_LSTM + j]);
      float gg = tanhf(gates[2 * D_LSTM + j]);
      float og = sigmoidf(gates[3 * D_LSTM + j]);
      cst = fg * cst + ig * gg;
      float hnew = og * tanhf(cst);
      hs[j] = hnew;
      outf[(size_t)t * D_LSTM + j] = hnew;
      outb[(size_t)t * D_LSTM + j] = (__bf16)hnew;
    }
    __syncthreads();
  }
}

// out[r] = lin2( relu( lin1(z[r]) ) ) ; one 128-thread block per row.
__global__ void mlp_head(const float* __restrict__ z, const float* __restrict__ l1w,
                         const float* __restrict__ l1b, const float* __restrict__ l2w,
                         const float* __restrict__ l2b, float* __restrict__ out) {
  __shared__ float row[D_LSTM];
  __shared__ float red[128];
  int r = blockIdx.x;
  int t = threadIdx.x;
  row[t]       = z[(size_t)r * D_LSTM + t];
  row[t + 128] = z[(size_t)r * D_LSTM + 128 + t];
  __syncthreads();
  float acc = l1b[t];
  const float* wrow = l1w + (size_t)t * D_LSTM;
#pragma unroll 8
  for (int k = 0; k < D_LSTM; ++k) acc += wrow[k] * row[k];
  acc = fmaxf(acc, 0.0f) * l2w[t];
  red[t] = acc;
  __syncthreads();
  for (int s = 64; s > 0; s >>= 1) {
    if (t < s) red[t] += red[t + s];
    __syncthreads();
  }
  if (t == 0) out[r] = red[0] + l2b[0];
}

// ---------------- host driver ----------------
extern "C" void kernel_launch(void* const* d_in, const int* in_sizes, int n_in,
                              void* d_out, int out_size, void* d_ws, size_t ws_size,
                              hipStream_t stream) {
  const float* x      = (const float*)d_in[0];
  const int*   eidx   = (const int*)d_in[1];
  const int*   etype  = (const int*)d_in[2];
  const int*   uidx   = (const int*)d_in[3];
  const int*   vidx   = (const int*)d_in[4];
  const float* basis  = (const float*)d_in[5];
  const float* comp   = (const float*)d_in[6];
  const float* root   = (const float*)d_in[7];
  const float* cbias  = (const float*)d_in[8];
  const float* w_ih   = (const float*)d_in[9];
  const float* w_hh   = (const float*)d_in[10];
  const float* b_ih   = (const float*)d_in[11];
  const float* b_hh   = (const float*)d_in[12];
  const float* lin1_w = (const float*)d_in[13];
  const float* lin1_b = (const float*)d_in[14];
  const float* lin2_w = (const float*)d_in[15];
  const float* lin2_b = (const float*)d_in[16];
  float* out = (float*)d_out;

  const int* src = eidx;
  const int* dst = eidx + N_EDGES;

  size_t off = 0;
  auto alloc = [&](size_t bytes) -> void* {
    void* p = (char*)d_ws + off;
    off += (bytes + 255) & ~(size_t)255;
    return p;
  };
  unsigned* counts = (unsigned*)alloc((size_t)N_NODES * N_REL * 4);
  float*  enorm    = (float*) alloc((size_t)N_EDGES * 4);
  __bf16* h_bf     = (__bf16*)alloc((size_t)N_NODES * H_DIM * 2);
  __bf16* Wb       = (__bf16*)alloc((size_t)N_LAYERS * (N_REL + 1) * H_DIM * H_DIM * 2);
  float*  y        = (float*) alloc((size_t)N_NODES * N_REL * H_DIM * 4);
  float*  agg      = (float*) alloc((size_t)N_NODES * H_DIM * 4);
  float*  states   = (float*) alloc((size_t)N_NODES * 128 * 4);
  __bf16* zb0      = (__bf16*)alloc((size_t)BATCH * D_LSTM * 2);
  __bf16* wih_bf   = (__bf16*)alloc((size_t)2 * G_LSTM * D_LSTM * 2);
  float*  xg       = (float*) alloc((size_t)BATCH * G_LSTM * 4);
  float*  zf1      = (float*) alloc((size_t)BATCH * D_LSTM * 4);
  __bf16* zb1      = (__bf16*)alloc((size_t)BATCH * D_LSTM * 2);
  float*  zf2      = (float*) alloc((size_t)BATCH * D_LSTM * 4);
  __bf16* zb2      = (__bf16*)alloc((size_t)BATCH * D_LSTM * 2);

  // --- edge mean-normalization ---
  hipMemsetAsync(counts, 0, (size_t)N_NODES * N_REL * 4, stream);
  edge_count<<<N_EDGES / 256, 256, 0, stream>>>(dst, etype, counts);
  edge_norm_k<<<N_EDGES / 256, 256, 0, stream>>>(dst, etype, counts, enorm);

  // --- prep: bf16 features + per-layer relation weight fragments + bf16 LSTM input weights ---
  cast_f32_bf16<<<(N_NODES * H_DIM) / 256, 256, 0, stream>>>(x, h_bf, N_NODES * H_DIM);
  build_relw<<<(N_LAYERS * (N_REL + 1) * H_DIM * H_DIM + 255) / 256, 256, 0, stream>>>(
      basis, comp, root, Wb);
  cast_f32_bf16<<<(2 * G_LSTM * D_LSTM) / 256, 256, 0, stream>>>(w_ih, wih_bf,
                                                                 2 * G_LSTM * D_LSTM);

  // --- RGCN layers ---
  for (int l = 0; l < N_LAYERS; ++l) {
    rgcn_transform<<<(N_NODES / 16) / 4, 128, 0, stream>>>(
        h_bf, Wb + (size_t)l * (N_REL + 1) * H_DIM * H_DIM, cbias + l * H_DIM, y, agg);
    rgcn_scatter<<<(N_EDGES * 8) / 256, 256, 0, stream>>>(src, dst, etype, enorm, y, agg);
    rgcn_update<<<(N_NODES * H_DIM) / 256, 256, 0, stream>>>(agg, states, h_bf, l);
  }

  // --- gather user/item states ---
  gather_z<<<(BATCH * D_LSTM) / 256, 256, 0, stream>>>(states, uidx, vidx, zb0);

  // --- LSTM layer 0 ---
  lstm_xgates<<<(BATCH / 16) * (G_LSTM / 16) / 4, 128, 0, stream>>>(
      zb0, wih_bf, b_ih, b_hh, xg);
  lstm_scan<<<1, 1024, 0, stream>>>(xg, w_hh, zf1, zb1);

  // --- LSTM layer 1 ---
  lstm_xgates<<<(BATCH / 16) * (G_LSTM / 16) / 4, 128, 0, stream>>>(
      zb1, wih_bf + (size_t)G_LSTM * D_LSTM, b_ih + G_LSTM, b_hh + G_LSTM, xg);
  lstm_scan<<<1, 1024, 0, stream>>>(xg, w_hh + (size_t)G_LSTM * D_LSTM, zf2, zb2);

  // --- MLP head ---
  mlp_head<<<BATCH, 128, 0, stream>>>(zf2, lin1_w, lin1_b, lin2_w, lin2_b, out);
}
